// TripleGCN_42314017800422
// MI455X (gfx1250) — matrix operand order repaired and checked by reading, not compiled, not run on hardware
//
#include <hip/hip_runtime.h>
#include <hip/hip_bf16.h>
#include <stdint.h>

#define N_NODES  100000
#define N_EDGES  1600000
#define IN_F     128
#define HID2     128
#define HID      64
#define OUT_F    64
#define N_GRAPHS 128

typedef __attribute__((ext_vector_type(16))) __bf16 v16bf;
typedef __attribute__((ext_vector_type(8)))  float  v8f;

union Frag16 {
  v16bf v;
  __bf16 h[16];
  unsigned short u[16];
  uint4 q[2];
};

// fp32 -> bf16 (RNE); compiler picks native cvt if the target has one
__device__ __forceinline__ unsigned short f2bf(float f) {
  return __builtin_bit_cast(unsigned short, static_cast<__bf16>(f));
}
// bf16 bits -> fp32 is a pure shift
__device__ __forceinline__ float bf2f_lo(unsigned int w) { return __uint_as_float(w << 16); }
__device__ __forceinline__ float bf2f_hi(unsigned int w) { return __uint_as_float(w & 0xFFFF0000u); }

// ---------------------------------------------------------------- zero fill
__global__ void tg_zero_f32(float* __restrict__ p, int n) {
  int i = blockIdx.x * blockDim.x + threadIdx.x;
  if (i < n) p[i] = 0.0f;
}

// ----------------------------------------------------- fp32 -> bf16 matrix
__global__ void tg_f32_to_bf16(const float* __restrict__ in, unsigned short* __restrict__ out,
                               int n) {
  int i = blockIdx.x * blockDim.x + threadIdx.x;
  if (i < n) out[i] = f2bf(in[i]);
}

// ------------------------------------------------- pack [Wr ; Ws] -> bf16
__global__ void tg_pack_w(const float* __restrict__ Wr, const float* __restrict__ Ws,
                          unsigned short* __restrict__ Wc, int dout, int K1) {
  int Kt = 2 * K1;
  int i = blockIdx.x * blockDim.x + threadIdx.x;
  if (i >= dout * Kt) return;
  int o = i / Kt;
  int k = i - o * Kt;
  float v = (k < K1) ? Wr[(size_t)o * K1 + k] : Ws[(size_t)o * K1 + (k - K1)];
  Wc[i] = f2bf(v);
}

// -------------------------------------- edge scatter: agg[dst] += h[src]
// h is bf16 (halves gather traffic); accumulation stays fp32 (L2 atomics).
// One thread per (edge, 4 features); VECS = d/4 is a power of two.
template <int VECS>
__global__ void tg_scatter_add_bf16(const unsigned short* __restrict__ h,
                                    const int* __restrict__ src, const int* __restrict__ dst,
                                    float* __restrict__ agg) {
  constexpr int D = VECS * 4;
  int i = blockIdx.x * blockDim.x + threadIdx.x;
  if (i >= N_EDGES * VECS) return;
  int e = i / VECS;
  int f = (i - e * VECS) << 2;
  int s = src[e];
  int t = dst[e];
  const uint2 raw = *reinterpret_cast<const uint2*>(h + (size_t)s * D + f);  // 4 x bf16
  float* a = agg + (size_t)t * D + f;
  atomicAdd(a + 0, bf2f_lo(raw.x));
  atomicAdd(a + 1, bf2f_hi(raw.x));
  atomicAdd(a + 2, bf2f_lo(raw.y));
  atomicAdd(a + 3, bf2f_hi(raw.y));
}

// ------------------------------------------------------------------------
// Fused GraphConv dense step:
//   Out = relu( [Agg(fp32) | Root(bf16)] (N x 2K1) * Wc^T (2K1 x dout) + b )
// One wave per 16x16 tile; K loop fully unrolled (8 or 4 WMMA per wave).
// Because K1 % 32 == 0, each unrolled iteration is statically either the
// Agg (fp32: load b128 x4 + cvt) side or the Root (bf16: load b128 x2) side.
// A (16-bit 16x32 layout): lane l = row M=l&15, half=l>>4,
//   element e -> K = k0 + e + 8*half + (e>=8 ? 8 : 0)  (two 8-wide runs).
// B (16-bit 32x16 layout): lane l = col N=l&15, same K mapping as A (two 8-wide runs).
// C/D: VGPR v, lane l -> M = v + 8*half, N = l&15.
// ------------------------------------------------------------------------
template <int K1, int DOUT, int TILESN>
__global__ void tg_gemm_bias_relu(const float* __restrict__ Agg,
                                  const unsigned short* __restrict__ Root,  // bf16 [N,K1]
                                  const unsigned short* __restrict__ Wc,    // bf16 [DOUT,2K1]
                                  const float* __restrict__ bias,
                                  unsigned short* __restrict__ Out) {       // bf16 [N,DOUT]
  constexpr int Kt = 2 * K1;
  const int wid  = threadIdx.x >> 5;
  const int lane = threadIdx.x & 31;
  const int wg   = blockIdx.x * (blockDim.x >> 5) + wid;
  const int tm   = wg / TILESN;       // TILESN is 8 or 4 -> shift
  const int tn   = wg - tm * TILESN;
  const int row  = lane & 15;
  const int half = lane >> 4;
  const int node = tm * 16 + row;     // N_NODES % 16 == 0: no guard, EXEC full
  const int n0   = tn * 16;

  const unsigned short* wrow = Wc + (size_t)(n0 + row) * Kt;

  v8f acc = {};
#pragma unroll
  for (int k0 = 0; k0 < Kt; k0 += 32) {
    Frag16 a;
    const int ka = k0 + 8 * half;
    const int kb = k0 + 16 + 8 * half;
    if (k0 < K1) {
      // both 8-wide runs fall in the fp32 Agg half
      const float* pa = Agg + (size_t)node * K1 + ka;
      const float* pb = Agg + (size_t)node * K1 + kb;
      float4 fa0 = reinterpret_cast<const float4*>(pa)[0];
      float4 fa1 = reinterpret_cast<const float4*>(pa)[1];
      float4 fb0 = reinterpret_cast<const float4*>(pb)[0];
      float4 fb1 = reinterpret_cast<const float4*>(pb)[1];
      a.h[0]  = (__bf16)fa0.x; a.h[1]  = (__bf16)fa0.y;
      a.h[2]  = (__bf16)fa0.z; a.h[3]  = (__bf16)fa0.w;
      a.h[4]  = (__bf16)fa1.x; a.h[5]  = (__bf16)fa1.y;
      a.h[6]  = (__bf16)fa1.z; a.h[7]  = (__bf16)fa1.w;
      a.h[8]  = (__bf16)fb0.x; a.h[9]  = (__bf16)fb0.y;
      a.h[10] = (__bf16)fb0.z; a.h[11] = (__bf16)fb0.w;
      a.h[12] = (__bf16)fb1.x; a.h[13] = (__bf16)fb1.y;
      a.h[14] = (__bf16)fb1.z; a.h[15] = (__bf16)fb1.w;
    } else {
      // both runs fall in the bf16 Root half: direct 16-byte loads
      const unsigned short* pa = Root + (size_t)node * K1 + (ka - K1);
      const unsigned short* pb = Root + (size_t)node * K1 + (kb - K1);
      a.q[0] = *reinterpret_cast<const uint4*>(pa);
      a.q[1] = *reinterpret_cast<const uint4*>(pb);
    }

    Frag16 b;
    const uint4* wp = reinterpret_cast<const uint4*>(wrow + k0 + 8 * half);
    b.q[0] = wp[0];
    b.q[1] = wp[2];

    acc = __builtin_amdgcn_wmma_f32_16x16x32_bf16(
        /*neg_a=*/false, a.v, /*neg_b=*/false, b.v,
        /*c_mod=*/(short)0, acc, /*reuse_a=*/false, /*reuse_b=*/false);
  }

  const float bn = bias[n0 + row];
#pragma unroll
  for (int v = 0; v < 8; ++v) {
    int m = tm * 16 + v + 8 * half;
    float val = acc[v] + bn;
    Out[(size_t)m * DOUT + n0 + row] = f2bf(val > 0.0f ? val : 0.0f);
  }
}

// --------------------------------------------------- graph readout (mean)
__global__ void tg_pool_accum(const unsigned short* __restrict__ h,  // bf16 [N, OUT_F]
                              const int* __restrict__ batch,
                              float* __restrict__ sums, float* __restrict__ counts) {
  int i = blockIdx.x * blockDim.x + threadIdx.x;
  if (i >= N_NODES * OUT_F) return;
  int node = i >> 6;          // OUT_F == 64
  int f = i & 63;
  int g = batch[node];
  atomicAdd(&sums[(size_t)g * OUT_F + f], bf2f_lo((unsigned int)h[i]));
  if (f == 0) atomicAdd(&counts[g], 1.0f);
}

__global__ void tg_pool_div(float* __restrict__ out, const float* __restrict__ counts) {
  int i = blockIdx.x * blockDim.x + threadIdx.x;
  if (i >= N_GRAPHS * OUT_F) return;
  float c = counts[i >> 6];
  out[i] = out[i] / (c > 1.0f ? c : 1.0f);
}

// ------------------------------------------------------------------------
extern "C" void kernel_launch(void* const* d_in, const int* in_sizes, int n_in,
                              void* d_out, int out_size, void* d_ws, size_t ws_size,
                              hipStream_t stream) {
  (void)in_sizes; (void)n_in; (void)out_size; (void)ws_size;

  const float* x    = (const float*)d_in[0];
  const int*   ei   = (const int*)d_in[1];
  const int*   bat  = (const int*)d_in[2];
  const float* W1r  = (const float*)d_in[3];
  const float* W1s  = (const float*)d_in[4];
  const float* b1   = (const float*)d_in[5];
  const float* W2r  = (const float*)d_in[6];
  const float* W2s  = (const float*)d_in[7];
  const float* b2   = (const float*)d_in[8];
  const float* W3r  = (const float*)d_in[9];
  const float* W3s  = (const float*)d_in[10];
  const float* b3   = (const float*)d_in[11];

  const int* src = ei;
  const int* dst = ei + N_EDGES;

  // workspace layout (bytes); total ~128.2 MB
  char* ws = (char*)d_ws;
  float*          agg = (float*)(ws + 0);                    // 100000*128*4 = 51,200,000
  unsigned short* xbf = (unsigned short*)(ws + 51200000);    // 100000*128*2 = 25,600,000
  unsigned short* h1  = (unsigned short*)(ws + 76800000);    // 25,600,000
  unsigned short* h2  = (unsigned short*)(ws + 102400000);   // 100000*64*2 = 12,800,000
  unsigned short* h3  = (unsigned short*)(ws + 115200000);   // 12,800,000
  unsigned short* Wc1 = (unsigned short*)(ws + 128000000);   // 128*256 bf16
  unsigned short* Wc2 = Wc1 + 128 * 256;                     // 64*256 bf16
  unsigned short* Wc3 = Wc2 + 64 * 256;                      // 64*128 bf16
  float* counts = (float*)(Wc3 + 64 * 128);                  // 128 f32
  float* out = (float*)d_out;

  // ---- one-time (per call) conversions: x and concatenated weights -> bf16
  tg_f32_to_bf16<<<(N_NODES * 128 + 255) / 256, 256, 0, stream>>>(x, xbf, N_NODES * 128);
  tg_pack_w<<<(128 * 256 + 255) / 256, 256, 0, stream>>>(W1r, W1s, Wc1, 128, 128);
  tg_pack_w<<<(64 * 256 + 255) / 256, 256, 0, stream>>>(W2r, W2s, Wc2, 64, 128);
  tg_pack_w<<<(64 * 128 + 255) / 256, 256, 0, stream>>>(W3r, W3s, Wc3, 64, 64);

  const int TM = N_NODES / 16;  // 6250 node tiles

  // ---- layer 1: 128 -> 128
  tg_zero_f32<<<(N_NODES * 128 + 255) / 256, 256, 0, stream>>>(agg, N_NODES * 128);
  tg_scatter_add_bf16<32><<<(N_EDGES * 32 + 255) / 256, 256, 0, stream>>>(xbf, src, dst, agg);
  tg_gemm_bias_relu<128, 128, 8><<<(TM * 8) / 4, 128, 0, stream>>>(agg, xbf, Wc1, b1, h1);

  // ---- layer 2: 128 -> 64
  tg_zero_f32<<<(N_NODES * 128 + 255) / 256, 256, 0, stream>>>(agg, N_NODES * 128);
  tg_scatter_add_bf16<32><<<(N_EDGES * 32 + 255) / 256, 256, 0, stream>>>(h1, src, dst, agg);
  tg_gemm_bias_relu<128, 64, 4><<<(TM * 4) / 4, 128, 0, stream>>>(agg, h1, Wc2, b2, h2);

  // ---- layer 3: 64 -> 64
  tg_zero_f32<<<(N_NODES * 64 + 255) / 256, 256, 0, stream>>>(agg, N_NODES * 64);
  tg_scatter_add_bf16<16><<<(N_EDGES * 16 + 255) / 256, 256, 0, stream>>>(h2, src, dst, agg);
  tg_gemm_bias_relu<64, 64, 4><<<(TM * 4) / 4, 128, 0, stream>>>(agg, h2, Wc3, b3, h3);

  // ---- scatter-mean readout
  tg_zero_f32<<<(N_GRAPHS * OUT_F + 255) / 256, 256, 0, stream>>>(out, N_GRAPHS * OUT_F);
  tg_zero_f32<<<1, 256, 0, stream>>>(counts, N_GRAPHS);
  tg_pool_accum<<<(N_NODES * OUT_F + 255) / 256, 256, 0, stream>>>(h3, bat, out, counts);
  tg_pool_div<<<(N_GRAPHS * OUT_F + 255) / 256, 256, 0, stream>>>(out, counts);
}
